// Spatial_87522843561664
// MI455X (gfx1250) — compile-verified, matrix operand order checked by reading, not run on hardware
//
#include <hip/hip_runtime.h>

// ---------------------------------------------------------------------------
// Types for CDNA5 WMMA (gfx1250, wave32)
// ---------------------------------------------------------------------------
typedef __attribute__((ext_vector_type(16))) __bf16 v16bf;
typedef __attribute__((ext_vector_type(8)))  float  v8f;
typedef __attribute__((ext_vector_type(4)))  unsigned int u32x4;

union FragU { v16bf v; u32x4 q[2]; };

// Load a 16x32 bf16 A/B fragment (lane = row/col index, K split per ISA table:
// lanes 0-15 get K = k0+0..7 and k0+16..23, lanes 16-31 get K = k0+8..15 and
// k0+24..31). Row-major source with leading dimension ldk (multiple of 8).
static __device__ inline v16bf load_frag(const __bf16* base, int ldk, int row,
                                         int k0, int half) {
  const __bf16* p = base + (size_t)row * ldk + k0 + half * 8;
  FragU f;
  f.q[0] = *(const u32x4*)(p);
  f.q[1] = *(const u32x4*)(p + 16);
  return f.v;
}

static __device__ inline v8f wmma_bf16(v16bf a, v16bf b, v8f c) {
  return __builtin_amdgcn_wmma_f32_16x16x32_bf16(false, a, false, b,
                                                 (short)0, c, false, false);
}

#define NLY 3
#define SEQ 24
#define DM  128
#define DFFN 512
#define LW  12
#define NINST 9600   // 16 * 600
#define NCH 600
#define NB 16

// ---------------------------------------------------------------------------
// f32 -> bf16 transposed weight convert: src is K x N, dst is N x K
// ---------------------------------------------------------------------------
__global__ void convT(const float* __restrict__ src, __bf16* __restrict__ dst,
                      int K, int N) {
  int gid = blockIdx.x * blockDim.x + threadIdx.x;
  if (gid >= K * N) return;
  int n = gid / K, k = gid % K;
  dst[gid] = (__bf16)src[k * N + n];
}

// ---------------------------------------------------------------------------
// Correlation + top-24 selection. One block per (b, i).
// ---------------------------------------------------------------------------
__global__ __launch_bounds__(256) void corr_topk(const float* __restrict__ x_pr,
                                                 const int* __restrict__ numberp,
                                                 int* __restrict__ idxout) {
  const int b = blockIdx.x / NCH;
  const int i = blockIdx.x % NCH;
  const int tid = threadIdx.x;
  __shared__ float sc[NCH];
  __shared__ float xci[LW];
  __shared__ float nrm_i;
  __shared__ float rbv[256];
  __shared__ int   rbi[256];

  const int num = *numberp;
  if (tid == 0) {
    float v[LW]; float m = 0.f;
    for (int l = 0; l < LW; ++l) {
      v[l] = x_pr[((b * LW + l) + num) * NCH + i];
      m += v[l];
    }
    m *= (1.f / LW);
    float nn = 0.f;
    for (int l = 0; l < LW; ++l) { v[l] -= m; nn += v[l] * v[l]; xci[l] = v[l]; }
    nrm_i = sqrtf(nn) + 1e-8f;
  }
  __syncthreads();

  for (int j = tid; j < NCH; j += 256) {
    float v[LW]; float m = 0.f;
    for (int l = 0; l < LW; ++l) {
      v[l] = x_pr[((b * LW + l) + num) * NCH + j];
      m += v[l];
    }
    m *= (1.f / LW);
    float dot = 0.f, nn = 0.f;
    for (int l = 0; l < LW; ++l) {
      float c = v[l] - m;
      dot += c * xci[l];
      nn += c * c;
    }
    sc[j] = dot / ((sqrtf(nn) + 1e-8f) * nrm_i);
  }
  __syncthreads();

  for (int k = 0; k < SEQ; ++k) {
    float best = -1e30f; int bi = NCH;
    for (int j = tid; j < NCH; j += 256) {
      float v = sc[j];
      if (v > best || (v == best && j < bi)) { best = v; bi = j; }
    }
    rbv[tid] = best; rbi[tid] = bi;
    __syncthreads();
    for (int s = 128; s > 0; s >>= 1) {
      if (tid < s) {
        if (rbv[tid + s] > rbv[tid] ||
            (rbv[tid + s] == rbv[tid] && rbi[tid + s] < rbi[tid])) {
          rbv[tid] = rbv[tid + s]; rbi[tid] = rbi[tid + s];
        }
      }
      __syncthreads();
    }
    if (tid == 0) {
      idxout[blockIdx.x * SEQ + k] = rbi[0];
      sc[rbi[0]] = -1e30f;
    }
    __syncthreads();
  }
}

// ---------------------------------------------------------------------------
// tgt preparation: tgt[b,c,l] = mean_r x_p[b,r,l,0,c]; also writes tgt_out.
// ---------------------------------------------------------------------------
__global__ void tgt_prep(const float* __restrict__ x_p,
                         const int* __restrict__ numberp,
                         float* __restrict__ tgt, float* __restrict__ tgt_out) {
  int gid = blockIdx.x * blockDim.x + threadIdx.x;
  if (gid >= NB * NCH * LW) return;
  int b = gid / (NCH * LW);
  int rem = gid % (NCH * LW);
  int c = rem / LW, l = rem % LW;
  int num = *numberp;
  float s = 0.f;
  for (int r = 0; r < 3; ++r)
    s += x_p[(((b * 3 + r) * LW + l) + num) * NCH + c];
  s *= (1.f / 3.f);
  tgt[gid] = s;
  tgt_out[(b * LW + l) * NCH + c] = s;
}

// ---------------------------------------------------------------------------
// Gather + source embedding: mem[inst,t,d] = sum_l xn[b,idx,l]*W[l,d] + b[d]
// ---------------------------------------------------------------------------
__global__ void enc_embed(const float* __restrict__ x_pr,
                          const int* __restrict__ numberp,
                          const int* __restrict__ idx,
                          const float* __restrict__ W,
                          const float* __restrict__ bias,
                          float* __restrict__ mem) {
  long long gid = (long long)blockIdx.x * blockDim.x + threadIdx.x;
  if (gid >= (long long)NINST * SEQ * DM) return;
  int d = (int)(gid & (DM - 1));
  int row = (int)(gid >> 7);
  int inst = row / SEQ, t = row % SEQ;
  int b = inst / NCH;
  int num = *numberp;
  int j = idx[inst * SEQ + t];
  float s = bias[d];
  for (int l = 0; l < LW; ++l)
    s += x_pr[((b * LW + l) + num) * NCH + j] * W[l * DM + d];
  mem[gid] = s;
}

// ---------------------------------------------------------------------------
// Target embedding: y[i,d] = sum_l tgt[i,l]*W[l,d] + b[d]
// ---------------------------------------------------------------------------
__global__ void tgt_embed(const float* __restrict__ tgt,
                          const float* __restrict__ W,
                          const float* __restrict__ bias,
                          float* __restrict__ y) {
  int gid = blockIdx.x * blockDim.x + threadIdx.x;
  if (gid >= NINST * DM) return;
  int d = gid & (DM - 1);
  int i = gid >> 7;
  float s = bias[d];
  for (int l = 0; l < LW; ++l) s += tgt[i * LW + l] * W[l * DM + d];
  y[gid] = s;
}

// ---------------------------------------------------------------------------
// Row LayerNorm (128 cols), f32 in -> bf16 out. One wave per row.
// ---------------------------------------------------------------------------
__global__ __launch_bounds__(256) void ln_rows(const float* __restrict__ X,
                                               __bf16* __restrict__ Y,
                                               int rows) {
  int wave = threadIdx.x >> 5, lane = threadIdx.x & 31;
  int row = blockIdx.x * 8 + wave;
  if (row >= rows) return;
  const float* x = X + (size_t)row * DM;
  float v0 = x[lane], v1 = x[lane + 32], v2 = x[lane + 64], v3 = x[lane + 96];
  float s = v0 + v1 + v2 + v3;
  float q = v0 * v0 + v1 * v1 + v2 * v2 + v3 * v3;
  for (int m = 16; m > 0; m >>= 1) {
    s += __shfl_xor(s, m, 32);
    q += __shfl_xor(q, m, 32);
  }
  float mean = s * (1.f / DM);
  float var = q * (1.f / DM) - mean * mean;
  float r = rsqrtf(var + 1e-6f);
  __bf16* y = Y + (size_t)row * DM;
  y[lane]      = (__bf16)((v0 - mean) * r);
  y[lane + 32] = (__bf16)((v1 - mean) * r);
  y[lane + 64] = (__bf16)((v2 - mean) * r);
  y[lane + 96] = (__bf16)((v3 - mean) * r);
}

// ---------------------------------------------------------------------------
// Generic WMMA GEMM: C = op(A(MxK,bf16) @ Bt(NxK,bf16) + bias [+ Cres])
// Block = 4 waves, each computing one 16x16 tile; grid = (M/16, N/64).
// ---------------------------------------------------------------------------
__global__ __launch_bounds__(128) void gemm_bf16(
    const __bf16* __restrict__ A, const __bf16* __restrict__ Bt,
    const float* __restrict__ bias, const float* __restrict__ Cres,
    float* __restrict__ Cf, __bf16* __restrict__ Cb,
    int M, int N, int K, int relu) {
  int wave = threadIdx.x >> 5, lane = threadIdx.x & 31;
  int half = lane >> 4, idx = lane & 15;
  int tm = blockIdx.x * 16;
  int tn = blockIdx.y * 64 + wave * 16;
  if (tm >= M || tn >= N) return;
  v8f acc = {0.f, 0.f, 0.f, 0.f, 0.f, 0.f, 0.f, 0.f};
  for (int k0 = 0; k0 < K; k0 += 32) {
    v16bf a = load_frag(A, K, tm + idx, k0, half);
    v16bf b = load_frag(Bt, K, tn + idx, k0, half);
    acc = wmma_bf16(a, b, acc);
  }
  int n = tn + idx;
  float bia = bias ? bias[n] : 0.f;
  for (int r = 0; r < 8; ++r) {
    int m = tm + r + half * 8;
    size_t off = (size_t)m * N + n;
    float v = acc[r] + bia;
    if (relu) v = fmaxf(v, 0.f);
    if (Cres) v += Cres[off];
    if (Cf) Cf[off] = v;
    if (Cb) Cb[off] = (__bf16)v;
  }
}

// ---------------------------------------------------------------------------
// Fused encoder MHA: per instance. mem += MHA(LN(mem)).  8 waves / 53KB LDS.
// ---------------------------------------------------------------------------
__global__ __launch_bounds__(256) void enc_mha(float* __restrict__ mem,
                                               const __bf16* __restrict__ WqT,
                                               const __bf16* __restrict__ WkT,
                                               const __bf16* __restrict__ WvT,
                                               const __bf16* __restrict__ WoT) {
  __shared__ float xs[SEQ * DM];
  __shared__ alignas(16) __bf16 hb[32 * DM];
  __shared__ alignas(16) __bf16 qb[32 * DM];
  __shared__ alignas(16) __bf16 kb[32 * DM];
  __shared__ alignas(16) __bf16 vb[32 * DM];
  __shared__ alignas(16) __bf16 ob[32 * DM];
  const int tid = threadIdx.x;
  const size_t base = (size_t)blockIdx.x * (SEQ * DM);

  for (int t = tid; t < SEQ * DM; t += 256) xs[t] = mem[base + t];
  for (int t = tid; t < 8 * DM; t += 256) {
    hb[SEQ * DM + t] = (__bf16)0.f;
    ob[SEQ * DM + t] = (__bf16)0.f;
  }
  __syncthreads();

  const int wave = tid >> 5, lane = tid & 31;
  // LayerNorm: wave w handles rows 3w..3w+2
  for (int rr = 0; rr < 3; ++rr) {
    int row = wave * 3 + rr;
    const float* x = xs + row * DM;
    float v0 = x[lane], v1 = x[lane + 32], v2 = x[lane + 64], v3 = x[lane + 96];
    float s = v0 + v1 + v2 + v3;
    float q = v0 * v0 + v1 * v1 + v2 * v2 + v3 * v3;
    for (int m = 16; m > 0; m >>= 1) {
      s += __shfl_xor(s, m, 32);
      q += __shfl_xor(q, m, 32);
    }
    float mean = s * (1.f / DM);
    float var = q * (1.f / DM) - mean * mean;
    float r = rsqrtf(var + 1e-6f);
    __bf16* y = hb + row * DM;
    y[lane]      = (__bf16)((v0 - mean) * r);
    y[lane + 32] = (__bf16)((v1 - mean) * r);
    y[lane + 64] = (__bf16)((v2 - mean) * r);
    y[lane + 96] = (__bf16)((v3 - mean) * r);
  }
  __syncthreads();

  const int half = lane >> 4, idx = lane & 15;
  // Q/K/V projections: 48 16x16 output tiles across 8 waves.
  for (int ti = wave; ti < 48; ti += 8) {
    int mat = ti >> 4, rem = ti & 15, mt = rem >> 3, nt = rem & 7;
    const __bf16* Wt = (mat == 0) ? WqT : (mat == 1 ? WkT : WvT);
    __bf16* dst = (mat == 0) ? qb : (mat == 1 ? kb : vb);
    v8f acc = {0.f, 0.f, 0.f, 0.f, 0.f, 0.f, 0.f, 0.f};
    for (int k0 = 0; k0 < DM; k0 += 32) {
      v16bf a = load_frag(hb, DM, mt * 16 + idx, k0, half);
      v16bf b = load_frag(Wt, DM, nt * 16 + idx, k0, half);
      acc = wmma_bf16(a, b, acc);
    }
    for (int r = 0; r < 8; ++r)
      dst[(mt * 16 + r + half * 8) * DM + nt * 16 + idx] = (__bf16)acc[r];
  }
  __syncthreads();

  // Attention: thread = (head, query). dh = 16, scale 1/4.
  if (tid < 8 * SEQ) {
    int hd = tid / SEQ, t = tid % SEQ;
    float qv[16];
    for (int d = 0; d < 16; ++d) qv[d] = (float)qb[t * DM + hd * 16 + d];
    float sc[SEQ]; float mx = -1e30f;
    for (int s = 0; s < SEQ; ++s) {
      float dot = 0.f;
      for (int d = 0; d < 16; ++d) dot += qv[d] * (float)kb[s * DM + hd * 16 + d];
      dot *= 0.25f;
      sc[s] = dot;
      mx = fmaxf(mx, dot);
    }
    float sum = 0.f;
    for (int s = 0; s < SEQ; ++s) { sc[s] = __expf(sc[s] - mx); sum += sc[s]; }
    float inv = 1.f / sum;
    for (int d = 0; d < 16; ++d) {
      float o = 0.f;
      for (int s = 0; s < SEQ; ++s) o += sc[s] * (float)vb[s * DM + hd * 16 + d];
      ob[t * DM + hd * 16 + d] = (__bf16)(o * inv);
    }
  }
  __syncthreads();

  // Output projection + residual: 16 tiles across 8 waves.
  for (int ti = wave; ti < 16; ti += 8) {
    int mt = ti >> 3, nt = ti & 7;
    v8f acc = {0.f, 0.f, 0.f, 0.f, 0.f, 0.f, 0.f, 0.f};
    for (int k0 = 0; k0 < DM; k0 += 32) {
      v16bf a = load_frag(ob, DM, mt * 16 + idx, k0, half);
      v16bf b = load_frag(WoT, DM, nt * 16 + idx, k0, half);
      acc = wmma_bf16(a, b, acc);
    }
    int n = nt * 16 + idx;
    for (int r = 0; r < 8; ++r) {
      int m = mt * 16 + r + half * 8;
      if (m < SEQ) mem[base + m * DM + n] = xs[m * DM + n] + acc[r];
    }
  }
}

// ---------------------------------------------------------------------------
// Fused encoder FFN: mem += relu(LN(mem)@W1+b1)@W2+b2. 53KB LDS.
// ---------------------------------------------------------------------------
__global__ __launch_bounds__(256) void enc_ffn(float* __restrict__ mem,
                                               const __bf16* __restrict__ W1T,
                                               const float* __restrict__ b1,
                                               const __bf16* __restrict__ W2T,
                                               const float* __restrict__ b2) {
  __shared__ float xs[SEQ * DM];
  __shared__ alignas(16) __bf16 hb[32 * DM];
  __shared__ alignas(16) __bf16 ub[32 * DFFN];
  const int tid = threadIdx.x;
  const size_t base = (size_t)blockIdx.x * (SEQ * DM);

  for (int t = tid; t < SEQ * DM; t += 256) xs[t] = mem[base + t];
  for (int t = tid; t < 8 * DM; t += 256) hb[SEQ * DM + t] = (__bf16)0.f;
  __syncthreads();

  const int wave = tid >> 5, lane = tid & 31;
  for (int rr = 0; rr < 3; ++rr) {
    int row = wave * 3 + rr;
    const float* x = xs + row * DM;
    float v0 = x[lane], v1 = x[lane + 32], v2 = x[lane + 64], v3 = x[lane + 96];
    float s = v0 + v1 + v2 + v3;
    float q = v0 * v0 + v1 * v1 + v2 * v2 + v3 * v3;
    for (int m = 16; m > 0; m >>= 1) {
      s += __shfl_xor(s, m, 32);
      q += __shfl_xor(q, m, 32);
    }
    float mean = s * (1.f / DM);
    float var = q * (1.f / DM) - mean * mean;
    float r = rsqrtf(var + 1e-6f);
    __bf16* y = hb + row * DM;
    y[lane]      = (__bf16)((v0 - mean) * r);
    y[lane + 32] = (__bf16)((v1 - mean) * r);
    y[lane + 64] = (__bf16)((v2 - mean) * r);
    y[lane + 96] = (__bf16)((v3 - mean) * r);
  }
  __syncthreads();

  const int half = lane >> 4, idx = lane & 15;
  // GEMM1: u = relu(h @ W1 + b1), 2x32 tiles.
  for (int ti = wave; ti < 64; ti += 8) {
    int mt = ti >> 5, nt = ti & 31;
    v8f acc = {0.f, 0.f, 0.f, 0.f, 0.f, 0.f, 0.f, 0.f};
    for (int k0 = 0; k0 < DM; k0 += 32) {
      v16bf a = load_frag(hb, DM, mt * 16 + idx, k0, half);
      v16bf b = load_frag(W1T, DM, nt * 16 + idx, k0, half);
      acc = wmma_bf16(a, b, acc);
    }
    int n = nt * 16 + idx;
    float bia = b1[n];
    for (int r = 0; r < 8; ++r) {
      int m = mt * 16 + r + half * 8;
      float v = fmaxf(acc[r] + bia, 0.f);
      ub[m * DFFN + n] = (__bf16)((m < SEQ) ? v : 0.f);
    }
  }
  __syncthreads();

  // GEMM2: mem = xs + u @ W2 + b2, 2x8 tiles, K=512.
  for (int ti = wave; ti < 16; ti += 8) {
    int mt = ti >> 3, nt = ti & 7;
    v8f acc = {0.f, 0.f, 0.f, 0.f, 0.f, 0.f, 0.f, 0.f};
    for (int k0 = 0; k0 < DFFN; k0 += 32) {
      v16bf a = load_frag(ub, DFFN, mt * 16 + idx, k0, half);
      v16bf b = load_frag(W2T, DFFN, nt * 16 + idx, k0, half);
      acc = wmma_bf16(a, b, acc);
    }
    int n = nt * 16 + idx;
    for (int r = 0; r < 8; ++r) {
      int m = mt * 16 + r + half * 8;
      if (m < SEQ) mem[base + m * DM + n] = xs[m * DM + n] + acc[r] + b2[n];
    }
  }
}

// ---------------------------------------------------------------------------
// Fused decoder cross-attention: y += CrossMHA(LN(y), memln).
// ---------------------------------------------------------------------------
__global__ __launch_bounds__(256) void dec_cross(float* __restrict__ y,
                                                 const __bf16* __restrict__ memln,
                                                 const float* __restrict__ cWq,
                                                 const __bf16* __restrict__ cWkT,
                                                 const __bf16* __restrict__ cWvT,
                                                 const float* __restrict__ cWo) {
  __shared__ float yr[DM], hq[DM], qv[DM], ov[DM];
  __shared__ float att[8 * SEQ];
  __shared__ float red[2];
  __shared__ alignas(16) __bf16 mb[32 * DM];
  __shared__ alignas(16) __bf16 kb[32 * DM];
  __shared__ alignas(16) __bf16 vb[32 * DM];
  const int tid = threadIdx.x;
  const size_t ybase = (size_t)blockIdx.x * DM;
  const size_t mbase = (size_t)blockIdx.x * (SEQ * DM);

  if (tid < DM) yr[tid] = y[ybase + tid];
  for (int t = tid; t < SEQ * DM; t += 256) mb[t] = memln[mbase + t];
  for (int t = tid; t < 8 * DM; t += 256) mb[SEQ * DM + t] = (__bf16)0.f;
  __syncthreads();

  const int wave = tid >> 5, lane = tid & 31;
  if (wave == 0) {
    float v0 = yr[lane], v1 = yr[lane + 32], v2 = yr[lane + 64], v3 = yr[lane + 96];
    float s = v0 + v1 + v2 + v3;
    float q = v0 * v0 + v1 * v1 + v2 * v2 + v3 * v3;
    for (int m = 16; m > 0; m >>= 1) {
      s += __shfl_xor(s, m, 32);
      q += __shfl_xor(q, m, 32);
    }
    float mean = s * (1.f / DM);
    float var = q * (1.f / DM) - mean * mean;
    if (lane == 0) { red[0] = mean; red[1] = rsqrtf(var + 1e-6f); }
  }
  __syncthreads();
  if (tid < DM) hq[tid] = (yr[tid] - red[0]) * red[1];

  const int half = lane >> 4, idx = lane & 15;
  // K/V = memln @ cWk / cWv : 32 tiles across 8 waves.
  for (int ti = wave; ti < 32; ti += 8) {
    int mat = ti >> 4, rem = ti & 15, mt = rem >> 3, nt = rem & 7;
    const __bf16* Wt = mat ? cWvT : cWkT;
    __bf16* dst = mat ? vb : kb;
    v8f acc = {0.f, 0.f, 0.f, 0.f, 0.f, 0.f, 0.f, 0.f};
    for (int k0 = 0; k0 < DM; k0 += 32) {
      v16bf a = load_frag(mb, DM, mt * 16 + idx, k0, half);
      v16bf b = load_frag(Wt, DM, nt * 16 + idx, k0, half);
      acc = wmma_bf16(a, b, acc);
    }
    for (int r = 0; r < 8; ++r)
      dst[(mt * 16 + r + half * 8) * DM + nt * 16 + idx] = (__bf16)acc[r];
  }
  __syncthreads();

  // q = hq @ cWq (1x128 row).
  if (tid < DM) {
    float s = 0.f;
    for (int k = 0; k < DM; ++k) s += hq[k] * cWq[k * DM + tid];
    qv[tid] = s;
  }
  __syncthreads();

  // Scores + softmax: one wave per head.
  {
    int hd = wave;
    float dot;
    if (lane < SEQ) {
      dot = 0.f;
      for (int d = 0; d < 16; ++d)
        dot += qv[hd * 16 + d] * (float)kb[lane * DM + hd * 16 + d];
      dot *= 0.25f;
    } else {
      dot = -1e30f;
    }
    float mx = dot;
    for (int m = 16; m > 0; m >>= 1) mx = fmaxf(mx, __shfl_xor(mx, m, 32));
    float e = (lane < SEQ) ? __expf(dot - mx) : 0.f;
    float sum = e;
    for (int m = 16; m > 0; m >>= 1) sum += __shfl_xor(sum, m, 32);
    if (lane < SEQ) att[hd * SEQ + lane] = e / sum;
  }
  __syncthreads();

  if (tid < DM) {
    int hd = tid >> 4;
    float o = 0.f;
    for (int s = 0; s < SEQ; ++s) o += att[hd * SEQ + s] * (float)vb[s * DM + tid];
    ov[tid] = o;
  }
  __syncthreads();

  if (tid < DM) {
    float s = 0.f;
    for (int k = 0; k < DM; ++k) s += ov[k] * cWo[k * DM + tid];
    y[ybase + tid] = yr[tid] + s;
  }
}

// ---------------------------------------------------------------------------
// Final generator projection: out[i,l] = yln[i,:] @ genW[:,l] + b[l]
// ---------------------------------------------------------------------------
__global__ void gen_proj(const __bf16* __restrict__ yln,
                         const float* __restrict__ W,
                         const float* __restrict__ b,
                         float* __restrict__ out) {
  int gid = blockIdx.x * blockDim.x + threadIdx.x;
  if (gid >= NINST * LW) return;
  int i = gid / LW, l = gid % LW;
  float s = b[l];
  for (int d = 0; d < DM; ++d) s += (float)yln[i * DM + d] * W[d * LW + l];
  out[gid] = s;
}

// ---------------------------------------------------------------------------
// Host launcher
// ---------------------------------------------------------------------------
extern "C" void kernel_launch(void* const* d_in, const int* in_sizes, int n_in,
                              void* d_out, int out_size, void* d_ws,
                              size_t ws_size, hipStream_t stream) {
  const float* x_pr = (const float*)d_in[0];
  const float* x_p  = (const float*)d_in[1];
  const int*   num  = (const int*)d_in[2];
  const float* srcW = (const float*)d_in[3];
  const float* srcB = (const float*)d_in[4];
  const float* tgtW = (const float*)d_in[5];
  const float* tgtB = (const float*)d_in[6];
  const float* encWq = (const float*)d_in[7];
  const float* encWk = (const float*)d_in[8];
  const float* encWv = (const float*)d_in[9];
  const float* encWo = (const float*)d_in[10];
  const float* encW1 = (const float*)d_in[11];
  const float* encB1 = (const float*)d_in[12];
  const float* encW2 = (const float*)d_in[13];
  const float* encB2 = (const float*)d_in[14];
  const float* sWv = (const float*)d_in[17];
  const float* sWo = (const float*)d_in[18];
  const float* cWq = (const float*)d_in[19];
  const float* cWk = (const float*)d_in[20];
  const float* cWv = (const float*)d_in[21];
  const float* cWo = (const float*)d_in[22];
  const float* dW1 = (const float*)d_in[23];
  const float* dB1 = (const float*)d_in[24];
  const float* dW2 = (const float*)d_in[25];
  const float* dB2 = (const float*)d_in[26];
  const float* genW = (const float*)d_in[27];
  const float* genB = (const float*)d_in[28];

  float* out_sq = (float*)d_out;
  float* out_tgt = out_sq + NINST * LW;

  // Workspace layout (256B aligned)
  char* ws = (char*)d_ws;
  __bf16* wts   = (__bf16*)(ws);                 // 1,179,648 bf16
  int*    idxb  = (int*)  (ws + 2359296);        // 9600*24 ints
  float*  tgt   = (float*)(ws + 3280896);        // 9600*12
  float*  mem   = (float*)(ws + 3741696);        // 230400*128 f32
  __bf16* memln = (__bf16*)(ws + 121706496);     // 230400*128 bf16
  float*  yb    = (float*)(ws + 180688896);      // 9600*128 f32
  __bf16* hb    = (__bf16*)(ws + 185604096);     // 9600*128 bf16
  __bf16* t1b   = (__bf16*)(ws + 188061696);     // 9600*128 bf16
  __bf16* ub    = (__bf16*)(ws + 190519296);     // 9600*512 bf16

  const size_t LYR = 196608;  // bf16 elems per layer weight block
  const size_t DECB = 3 * LYR;

  auto cvt = [&](const float* s, __bf16* d, int K, int N) {
    int n = K * N;
    convT<<<dim3((n + 255) / 256), dim3(256), 0, stream>>>(s, d, K, N);
  };

  // Transposed bf16 weights
  for (int i = 0; i < NLY; ++i) {
    size_t eb = (size_t)i * LYR;
    cvt(encWq + i * 16384, wts + eb + 0,      128, 128);
    cvt(encWk + i * 16384, wts + eb + 16384,  128, 128);
    cvt(encWv + i * 16384, wts + eb + 32768,  128, 128);
    cvt(encWo + i * 16384, wts + eb + 49152,  128, 128);
    cvt(encW1 + i * 65536, wts + eb + 65536,  128, 512);
    cvt(encW2 + i * 65536, wts + eb + 131072, 512, 128);
    size_t db = DECB + (size_t)i * LYR;
    cvt(sWv + i * 16384, wts + db + 0,      128, 128);
    cvt(sWo + i * 16384, wts + db + 16384,  128, 128);
    cvt(cWk + i * 16384, wts + db + 32768,  128, 128);
    cvt(cWv + i * 16384, wts + db + 49152,  128, 128);
    cvt(dW1 + i * 65536, wts + db + 65536,  128, 512);
    cvt(dW2 + i * 65536, wts + db + 131072, 512, 128);
  }

  // Correlation top-K, target prep, embeddings
  corr_topk<<<dim3(NINST), dim3(256), 0, stream>>>(x_pr, num, idxb);
  tgt_prep<<<dim3((NINST * LW + 255) / 256), dim3(256), 0, stream>>>(
      x_p, num, tgt, out_tgt);
  enc_embed<<<dim3((NINST * SEQ * DM) / 256), dim3(256), 0, stream>>>(
      x_pr, num, idxb, srcW, srcB, mem);

  // Encoder layers (fused per-instance MHA + FFN)
  for (int i = 0; i < NLY; ++i) {
    size_t eb = (size_t)i * LYR;
    enc_mha<<<dim3(NINST), dim3(256), 0, stream>>>(
        mem, wts + eb + 0, wts + eb + 16384, wts + eb + 32768, wts + eb + 49152);
    enc_ffn<<<dim3(NINST), dim3(256), 0, stream>>>(
        mem, wts + eb + 65536, encB1 + i * 512, wts + eb + 131072,
        encB2 + i * 128);
  }
  ln_rows<<<dim3(NINST * SEQ / 8), dim3(256), 0, stream>>>(mem, memln,
                                                           NINST * SEQ);

  // Decoder
  tgt_embed<<<dim3((NINST * DM) / 256), dim3(256), 0, stream>>>(tgt, tgtW, tgtB,
                                                                yb);
  for (int i = 0; i < NLY; ++i) {
    size_t db = DECB + (size_t)i * LYR;
    // Self-attn on 1 token reduces to y += LN(y) @ Wv @ Wo
    ln_rows<<<dim3(NINST / 8), dim3(256), 0, stream>>>(yb, hb, NINST);
    gemm_bf16<<<dim3(NINST / 16, 2), dim3(128), 0, stream>>>(
        hb, wts + db + 0, (const float*)nullptr, (const float*)nullptr,
        (float*)nullptr, t1b, NINST, 128, 128, 0);
    gemm_bf16<<<dim3(NINST / 16, 2), dim3(128), 0, stream>>>(
        t1b, wts + db + 16384, (const float*)nullptr, yb, yb, (__bf16*)nullptr,
        NINST, 128, 128, 0);
    // Cross attention (fused per instance)
    dec_cross<<<dim3(NINST), dim3(256), 0, stream>>>(
        yb, memln, cWq + i * 16384, wts + db + 32768, wts + db + 49152,
        cWo + i * 16384);
    // FFN
    ln_rows<<<dim3(NINST / 8), dim3(256), 0, stream>>>(yb, hb, NINST);
    gemm_bf16<<<dim3(NINST / 16, 8), dim3(128), 0, stream>>>(
        hb, wts + db + 65536, dB1 + i * 512, (const float*)nullptr,
        (float*)nullptr, ub, NINST, 512, 128, 1);
    gemm_bf16<<<dim3(NINST / 16, 2), dim3(128), 0, stream>>>(
        ub, wts + db + 131072, dB2 + i * 128, yb, yb, (__bf16*)nullptr, NINST,
        128, 512, 0);
  }

  // Final LN + generator projection
  ln_rows<<<dim3(NINST / 8), dim3(256), 0, stream>>>(yb, hb, NINST);
  gen_proj<<<dim3((NINST * LW + 255) / 256), dim3(256), 0, stream>>>(
      hb, genW, genB, out_sq);

  (void)in_sizes; (void)n_in; (void)out_size; (void)ws_size;
}
